// ResNetLoop_73203422593214
// MI455X (gfx1250) — compile-verified, matrix operand order checked by reading, not run on hardware
//
#include <hip/hip_runtime.h>
#include <math.h>

// ---------------------------------------------------------------------------
// Types for CDNA5 WMMA
// ---------------------------------------------------------------------------
typedef __attribute__((ext_vector_type(16))) __bf16 v16bf;
typedef __attribute__((ext_vector_type(8)))  float  v8f;

// bf16 <-> f32 helpers (bit level; RNE rounding)
__device__ __forceinline__ unsigned short f2bf(float f) {
    unsigned u = __float_as_uint(f);
    unsigned r = u + 0x7FFFu + ((u >> 16) & 1u);
    return (unsigned short)(r >> 16);
}
__device__ __forceinline__ float bf2f(unsigned short s) {
    return __uint_as_float(((unsigned)s) << 16);
}

#define EPSBN 1e-5f

// 32-byte fragment union: two 16B vector loads <-> one v16bf WMMA operand
union Frag {
    uint4          x[2];
    unsigned short u[16];
    v16bf          v;
};

// ---------------------------------------------------------------------------
// NCHW f32 -> NHWC bf16
// ---------------------------------------------------------------------------
__global__ __launch_bounds__(256) void nchw2nhwc_kernel(
    const float* __restrict__ x, unsigned short* __restrict__ out,
    int N, int C, int H, int W)
{
    long total = (long)N * C * H * W;
    long i = (long)blockIdx.x * blockDim.x + threadIdx.x;
    if (i >= total) return;
    int w = (int)(i % W); long t = i / W;
    int h = (int)(t % H); t /= H;
    int c = (int)(t % C); int n = (int)(t / C);
    out[(((long)n * H + h) * W + w) * C + c] = f2bf(x[i]);
}

// ---------------------------------------------------------------------------
// Pack OIHW f32 weights directly into WMMA B-fragment order:
//   ushort index = (((nt*nChunks + chunk)*32 + lane)*16 + j)
//   j in [0,8):  k = chunk*32 +      (lane>>4)*8 + j
//   j in [8,16): k = chunk*32 + 16 + (lane>>4)*8 + (j-8)
//   column      = nt*16 + (lane&15)
// so the conv kernels load B as two consecutive uint4 per (chunk,lane).
// ---------------------------------------------------------------------------
__global__ __launch_bounds__(256) void wpack_kernel(
    const float* __restrict__ w, unsigned short* __restrict__ out,
    int O, int I, int KH, int KW, int nChunks)
{
    long total = (long)(O >> 4) * nChunks * 32 * 16;
    long idx = (long)blockIdx.x * blockDim.x + threadIdx.x;
    if (idx >= total) return;
    int j = (int)(idx & 15); long t = idx >> 4;
    int lane = (int)(t & 31); t >>= 5;
    int chunk = (int)(t % nChunks);
    int nt    = (int)(t / nChunks);
    int khalf = lane >> 4;
    int col   = nt * 16 + (lane & 15);
    int kk = chunk * 32 + ((j >> 3) << 4) + (khalf << 3) + (j & 7);
    int KTOT = KH * KW * I;
    unsigned short v = 0;
    if (kk < KTOT) {
        int cin = kk % I;
        int r   = kk / I;
        int kw  = r % KW;
        int kh  = r / KW;
        v = f2bf(w[(((long)col * I + cin) * KH + kh) * KW + kw]);
    }
    out[idx] = v;
}

// ---------------------------------------------------------------------------
// Fast implicit-GEMM conv (requires Cin % 32 == 0, Cout % 64 == 0).
// Register-blocked: each wave computes a 64x64 output tile
// (4 M-subtiles x 4 N-subtiles = 16 accumulators).  Per cin-chunk:
// 8 aligned b128 A-loads + 8 b128 B-loads feed 16 back-to-back WMMAs,
// i.e. one b128 load per WMMA (~16 FLOP/byte from cache).
// ---------------------------------------------------------------------------
__global__ __launch_bounds__(256) void conv_fast_kernel(
    const unsigned short* __restrict__ in,
    const uint4* __restrict__ wp,
    float* __restrict__ out,
    int N, int H, int W, int Cin,
    int OH, int OW, int Cout,
    int KH, int KW, int stride, int pad, int nChunks)
{
    const int waveId = threadIdx.x >> 5;
    const int lane   = threadIdx.x & 31;
    const int M      = N * OH * OW;
    const int mTiles = (M + 63) >> 6;     // 64 rows per wave
    const int nTiles4 = Cout >> 6;        // 64 cols per wave
    long tile = (long)blockIdx.x * (blockDim.x >> 5) + waveId;
    if (tile >= (long)mTiles * nTiles4) return;
    const int mt  = (int)(tile / nTiles4);
    const int nt4 = (int)(tile % nTiles4);

    const int row   = lane & 15;
    const int khalf = lane >> 4;

    // four M rows handled per lane position
    int  nn[4], ohh[4], oww[4];
    bool vv[4];
    #pragma unroll
    for (int r = 0; r < 4; ++r) {
        int gm = mt * 64 + r * 16 + row;
        vv[r] = gm < M;
        int g = vv[r] ? gm : M - 1;
        nn[r]  = g / (OH * OW);
        int q  = g % (OH * OW);
        ohh[r] = q / OW;
        oww[r] = q % OW;
    }

    const int chunkPer = Cin >> 5;
    const long tstr = (long)nChunks * 64;          // uint4 stride between nt
    const uint4 zero4 = make_uint4(0u, 0u, 0u, 0u);

    v8f acc[16] = {};                               // acc[4*t + r]
    for (int kh = 0; kh < KH; ++kh) {
        for (int kw = 0; kw < KW; ++kw) {
            bool ok[4];
            const unsigned short* rb[4];
            #pragma unroll
            for (int r = 0; r < 4; ++r) {
                int ih = ohh[r] * stride - pad + kh;
                int iw = oww[r] * stride - pad + kw;
                ok[r] = vv[r] && (unsigned)ih < (unsigned)H
                              && (unsigned)iw < (unsigned)W;
                rb[r] = in + (((long)nn[r] * H + ih) * W + iw) * Cin;
            }
            int chunkBase = (kh * KW + kw) * chunkPer;
            for (int cc = 0; cc < chunkPer; ++cc) {
                Frag A[4];
                #pragma unroll
                for (int r = 0; r < 4; ++r) {
                    if (ok[r]) {
                        A[r].x[0] = *(const uint4*)(rb[r] + cc * 32 + khalf * 8);
                        A[r].x[1] = *(const uint4*)(rb[r] + cc * 32 + 16 + khalf * 8);
                    } else { A[r].x[0] = zero4; A[r].x[1] = zero4; }
                }
                const uint4* bp = wp
                    + ((long)(nt4 * 4) * nChunks + (chunkBase + cc)) * 64
                    + lane * 2;
                #pragma unroll
                for (int t = 0; t < 4; ++t) {
                    Frag B;
                    B.x[0] = bp[t * tstr];
                    B.x[1] = bp[t * tstr + 1];
                    // 4 independent WMMAs sharing this B fragment
                    #pragma unroll
                    for (int r = 0; r < 4; ++r) {
                        acc[4 * t + r] = __builtin_amdgcn_wmma_f32_16x16x32_bf16(
                            false, A[r].v, false, B.v, (short)0,
                            acc[4 * t + r], false, false);
                    }
                }
            }
        }
    }
    #pragma unroll
    for (int t = 0; t < 4; ++t) {
        int oc = (nt4 * 4 + t) * 16 + row;
        #pragma unroll
        for (int r = 0; r < 4; ++r) {
            #pragma unroll
            for (int j = 0; j < 8; ++j) {
                int om = mt * 64 + r * 16 + j + 8 * khalf;
                if (om < M) out[(long)om * Cout + oc] = acc[4 * t + r][j];
            }
        }
    }
}

// ---------------------------------------------------------------------------
// conv1 specialization: Cin=5, 7x7, stride 2, pad 3.  A-gather is scalar
// masked u16 loads but all k->(kh,kw,cin) decode uses compile-time divisors.
// B comes from the packed-fragment buffer (two b128 loads).
// ---------------------------------------------------------------------------
__global__ __launch_bounds__(256) void conv1_kernel(
    const unsigned short* __restrict__ in,
    const uint4* __restrict__ wp,
    float* __restrict__ out,
    int N, int H, int W, int OH, int OW, int Cout)
{
    constexpr int CIN = 5, KH = 7, KW = 7, STRIDE = 2, PAD = 3;
    constexpr int KTOT = KH * KW * CIN;           // 245
    constexpr int NCHUNK = (KTOT + 31) / 32;      // 8

    const int waveId = threadIdx.x >> 5;
    const int lane   = threadIdx.x & 31;
    const int M      = N * OH * OW;
    const int mTiles = (M + 15) >> 4;
    const int nTiles = Cout >> 4;
    long tile = (long)blockIdx.x * (blockDim.x >> 5) + waveId;
    if (tile >= (long)mTiles * nTiles) return;
    const int mt = (int)(tile / nTiles);
    const int nt = (int)(tile % nTiles);

    const int row   = lane & 15;
    const int khalf = lane >> 4;

    int gm = mt * 16 + row;
    bool rowValid = gm < M;
    int gmc = rowValid ? gm : M - 1;
    const int n  = gmc / (OH * OW);
    const int r2 = gmc % (OH * OW);
    const int oh = r2 / OW;
    const int ow = r2 % OW;

    v8f acc = {};
    for (int chunk = 0; chunk < NCHUNK; ++chunk) {
        Frag A, B;
        #pragma unroll
        for (int j = 0; j < 16; ++j) {
            int k = chunk * 32 + ((j >> 3) << 4) + (khalf << 3) + (j & 7);
            unsigned short av = 0;
            if (rowValid && k < KTOT) {
                int cin = k % CIN;                 // compile-time divisor
                int kk  = k / CIN;
                int kw  = kk % KW;
                int kh  = kk / KW;
                int ih  = oh * STRIDE - PAD + kh;
                int iw  = ow * STRIDE - PAD + kw;
                if ((unsigned)ih < (unsigned)H && (unsigned)iw < (unsigned)W)
                    av = in[(((long)n * H + ih) * W + iw) * CIN + cin];
            }
            A.u[j] = av;
        }
        const uint4* bp = wp + (((long)nt * NCHUNK + chunk) * 32 + lane) * 2;
        B.x[0] = bp[0];
        B.x[1] = bp[1];
        acc = __builtin_amdgcn_wmma_f32_16x16x32_bf16(
            false, A.v, false, B.v, (short)0, acc, false, false);
    }
    #pragma unroll
    for (int j = 0; j < 8; ++j) {
        int om = mt * 16 + j + 8 * khalf;
        if (om < M) out[(long)om * Cout + (nt * 16 + row)] = acc[j];
    }
}

// ---------------------------------------------------------------------------
// Per-channel batch statistics over [M, C] f32 (training-mode BN)
// ---------------------------------------------------------------------------
__global__ __launch_bounds__(256) void bn_stats_kernel(
    const float* __restrict__ x, long M, int C,
    float* __restrict__ mean, float* __restrict__ var)
{
    __shared__ float ssum[256], ssq[256];
    int c = blockIdx.x;
    float s = 0.f, q = 0.f;
    for (long m = threadIdx.x; m < M; m += blockDim.x) {
        float v = x[m * C + c];
        s += v; q += v * v;
    }
    ssum[threadIdx.x] = s; ssq[threadIdx.x] = q;
    __syncthreads();
    for (int st = 128; st > 0; st >>= 1) {
        if ((int)threadIdx.x < st) {
            ssum[threadIdx.x] += ssum[threadIdx.x + st];
            ssq[threadIdx.x]  += ssq[threadIdx.x + st];
        }
        __syncthreads();
    }
    if (threadIdx.x == 0) {
        float mu = ssum[0] / (float)M;
        mean[c] = mu;
        var[c]  = ssq[0] / (float)M - mu * mu;   // biased variance
    }
}

// ---------------------------------------------------------------------------
// BN normalize (+optional residual bf16) (+optional ReLU) -> bf16
// ---------------------------------------------------------------------------
__global__ __launch_bounds__(256) void bn_apply_kernel(
    const float* __restrict__ x,
    const float* __restrict__ mean, const float* __restrict__ var,
    const float* __restrict__ g, const float* __restrict__ b,
    const unsigned short* __restrict__ res, int relu,
    unsigned short* __restrict__ out, long total, int C)
{
    long i = (long)blockIdx.x * blockDim.x + threadIdx.x;
    if (i >= total) return;
    int c = (int)(i % C);
    float y = (x[i] - mean[c]) * rsqrtf(var[c] + EPSBN) * g[c] + b[c];
    if (res) y += bf2f(res[i]);
    if (relu) y = fmaxf(y, 0.f);
    out[i] = f2bf(y);
}

// ---------------------------------------------------------------------------
// 3x3 s2 p1 maxpool, NHWC bf16 (-inf padding)
// ---------------------------------------------------------------------------
__global__ __launch_bounds__(256) void maxpool_kernel(
    const unsigned short* __restrict__ in, unsigned short* __restrict__ out,
    int N, int H, int W, int C, int OH, int OW)
{
    long total = (long)N * OH * OW * C;
    long i = (long)blockIdx.x * blockDim.x + threadIdx.x;
    if (i >= total) return;
    int c  = (int)(i % C); long t = i / C;
    int ow = (int)(t % OW); t /= OW;
    int oh = (int)(t % OH); int n = (int)(t / OH);
    float m = -INFINITY;
    for (int kh = 0; kh < 3; ++kh) {
        int ih = oh * 2 - 1 + kh;
        if (ih < 0 || ih >= H) continue;
        for (int kw = 0; kw < 3; ++kw) {
            int iw = ow * 2 - 1 + kw;
            if (iw < 0 || iw >= W) continue;
            m = fmaxf(m, bf2f(in[(((long)n * H + ih) * W + iw) * C + c]));
        }
    }
    out[i] = f2bf(m);
}

// ---------------------------------------------------------------------------
// Aggregate helpers (corner scatter / gather + batch mean)
// ---------------------------------------------------------------------------
__global__ void agg_count_kernel(const int* __restrict__ pairs, int P,
                                 float* __restrict__ cnt, int NC)
{
    int t = threadIdx.x;
    if (t < NC) {
        float c = 0.f;
        for (int p = 0; p < P; ++p)
            if (pairs[2 * p] == t) c += 1.f;
        cnt[t] = fmaxf(c, 1.f);   // clip(count, 1)
    }
}

__global__ __launch_bounds__(256) void agg_scatter_kernel(
    const unsigned short* __restrict__ x, const int* __restrict__ pairs,
    float* __restrict__ csum, long HWC, int P)
{
    long total = (long)P * HWC;
    long i = (long)blockIdx.x * blockDim.x + threadIdx.x;
    if (i >= total) return;
    int p = (int)(i / HWC);
    long j = i % HWC;
    int corner = pairs[2 * p];
    int edge   = pairs[2 * p + 1];
    atomicAdd(&csum[(long)corner * HWC + j], bf2f(x[(long)edge * HWC + j]));
}

__global__ __launch_bounds__(256) void agg_mean_kernel(
    const unsigned short* __restrict__ x, float* __restrict__ meanbuf,
    long HWC, int E)
{
    long j = (long)blockIdx.x * blockDim.x + threadIdx.x;
    if (j >= HWC) return;
    float s = 0.f;
    for (int e = 0; e < E; ++e) s += bf2f(x[(long)e * HWC + j]);
    meanbuf[j] = s / (float)E;
}

__global__ __launch_bounds__(256) void agg_apply_kernel(
    unsigned short* __restrict__ x, const float* __restrict__ csum,
    const float* __restrict__ cnt, const float* __restrict__ meanbuf,
    const int* __restrict__ ec, long HWC, int E)
{
    long total = (long)E * HWC;
    long i = (long)blockIdx.x * blockDim.x + threadIdx.x;
    if (i >= total) return;
    int e = (int)(i / HWC);
    long j = i % HWC;
    int c0 = ec[2 * e], c1 = ec[2 * e + 1];
    float v = bf2f(x[i])
            + csum[(long)c0 * HWC + j] / cnt[c0]
            + csum[(long)c1 * HWC + j] / cnt[c1]
            + meanbuf[j];
    x[i] = f2bf(v);
}

// ---------------------------------------------------------------------------
// Adaptive avg pool  (NHWC bf16 [E,HW,C] -> f32 [E,C])
// ---------------------------------------------------------------------------
__global__ __launch_bounds__(256) void avgpool_kernel(
    const unsigned short* __restrict__ x, float* __restrict__ pooled,
    int E, int HW, int C)
{
    int idx = blockIdx.x * blockDim.x + threadIdx.x;
    if (idx >= E * C) return;
    int e = idx / C, c = idx % C;
    float s = 0.f;
    for (int h = 0; h < HW; ++h) s += bf2f(x[((long)e * HW + h) * C + c]);
    pooled[idx] = s / (float)HW;
}

// ---------------------------------------------------------------------------
// FC layers
// ---------------------------------------------------------------------------
__global__ __launch_bounds__(256) void fc1_kernel(
    const float* __restrict__ pooled, const float* __restrict__ w,
    const float* __restrict__ b, float* __restrict__ y,
    int E, int K, int J)
{
    int idx = blockIdx.x * blockDim.x + threadIdx.x;
    if (idx >= E * J) return;
    int e = idx / J, j = idx % J;
    float s = b[j];
    for (int k = 0; k < K; ++k) s += pooled[(long)e * K + k] * w[(long)j * K + k];
    y[(long)e * J + j] = s;
}

__global__ void fc2_kernel(
    const float* __restrict__ y, const float* __restrict__ w,
    const float* __restrict__ b, float* __restrict__ pred, int E, int J)
{
    int e = threadIdx.x;
    if (e >= E) return;
    float s = b[0];
    for (int j = 0; j < J; ++j) s += fmaxf(y[(long)e * J + j], 0.f) * w[j];
    pred[e] = 1.f / (1.f + expf(-s));
}

// ---------------------------------------------------------------------------
// Host orchestration
// ---------------------------------------------------------------------------
extern "C" void kernel_launch(void* const* d_in, const int* in_sizes, int n_in,
                              void* d_out, int out_size, void* d_ws, size_t ws_size,
                              hipStream_t stream)
{
    const int E = 64, NC = 48;
    const float* x_nchw = (const float*)d_in[0];
    const int*   pairs  = (const int*)d_in[1];   // [128, 2] (corner, edge)
    const int*   ec     = (const int*)d_in[2];   // [64, 2]
    // d_in[3] = num_corners scalar (known = 48)

    // ---- workspace carve-up -------------------------------------------------
    char* ws = (char*)d_ws;
    size_t off = 0;
    auto walloc = [&](size_t bytes) -> void* {
        void* p = ws + off;
        off = (off + bytes + 255) & ~(size_t)255;
        return p;
    };
    unsigned short* d_xin   = (unsigned short*)walloc((size_t)E * 256 * 256 * 5 * 2);
    unsigned short* d_wts   = (unsigned short*)walloc((size_t)24 * 1024 * 1024 * 2);
    float*          d_cf32  = (float*)walloc((size_t)68 * 1024 * 1024 * 4);
    unsigned short* d_actA  = (unsigned short*)walloc((size_t)68 * 1024 * 1024 * 2);
    unsigned short* d_actB  = (unsigned short*)walloc((size_t)17 * 1024 * 1024 * 2);
    unsigned short* d_actC  = (unsigned short*)walloc((size_t)17 * 1024 * 1024 * 2);
    unsigned short* d_actR  = (unsigned short*)walloc((size_t)9  * 1024 * 1024 * 2);
    float*          d_mean  = (float*)walloc(512 * 4);
    float*          d_var   = (float*)walloc(512 * 4);
    float*          d_cnt   = (float*)walloc(64 * 4);
    float*          d_csum  = (float*)walloc((size_t)NC * 262144 * 4);
    float*          d_mbuf  = (float*)walloc((size_t)262144 * 4);
    float*          d_pool  = (float*)walloc((size_t)E * 512 * 4);

    // ---- unpack params (jax tree_flatten order: dict keys sorted) ----------
    int pi = 4;
    auto nextp = [&]() -> const float* { return (const float*)d_in[pi++]; };
    const float* P_b1    = nextp();   // 'b1'
    const float* P_c1w   = nextp();   // 'conv1'
    const float* P_fc1b  = nextp();   // 'fc1_b'
    const float* P_fc1w  = nextp();   // 'fc1_w'
    const float* P_fc2b  = nextp();   // 'fc2_b'
    const float* P_fc2w  = nextp();   // 'fc2_w'
    const float* P_g1    = nextp();   // 'g1'

    size_t woff = 0;
    auto pack = [&](const float* w, int O, int I, int K) -> const unsigned short* {
        int KTOT = K * K * I;
        int nChunks = (KTOT + 31) / 32;
        unsigned short* dst = d_wts + woff;
        size_t total = (size_t)(O >> 4) * nChunks * 512;
        woff += (total + 127) & ~(size_t)127;
        int blocks = (int)((total + 255) / 256);
        wpack_kernel<<<blocks, 256, 0, stream>>>(w, dst, O, I, K, K, nChunks);
        return dst;
    };

    const unsigned short* W_conv1 = pack(P_c1w, 64, 5, 7);

    struct BP {
        const float *g1, *b1, *g2, *b2;
        const unsigned short *w1, *w2;
        bool down;
        const float *dg, *db;
        const unsigned short *dw;
        int stride, inpl, planes;
    };
    static const int SPECS[4][3] = {{64,3,1},{128,4,2},{256,6,2},{512,3,2}};
    BP blocks[16];
    int nb = 0, inpl = 64;
    for (int L = 0; L < 4; ++L) {
        int planes = SPECS[L][0], nblocks = SPECS[L][1], stride = SPECS[L][2];
        for (int bi = 0; bi < nblocks; ++bi) {
            int s = (bi == 0) ? stride : 1;
            bool down = (s != 1) || (inpl != planes);
            BP bp{};
            // sorted block keys: b1, b2, conv1, conv2, [down.b, down.g, down.w], g1, g2
            bp.b1 = nextp();
            bp.b2 = nextp();
            const float* c1 = nextp();
            const float* c2 = nextp();
            if (down) {
                bp.db = nextp();
                bp.dg = nextp();
                bp.dw = pack(nextp(), planes, inpl, 1);
            }
            bp.g1 = nextp();
            bp.g2 = nextp();
            bp.w1 = pack(c1, planes, inpl, 3);
            bp.w2 = pack(c2, planes, planes, 3);
            bp.down = down; bp.stride = s; bp.inpl = inpl; bp.planes = planes;
            blocks[nb++] = bp;
            inpl = planes;
        }
    }

    // ---- small launch helpers ----------------------------------------------
    auto conv = [&](const unsigned short* in, const unsigned short* wp, float* out,
                    int N, int H, int W, int Cin, int Cout,
                    int KH, int KW, int s, int pad, int& OH, int& OW) {
        OH = (H + 2 * pad - KH) / s + 1;
        OW = (W + 2 * pad - KW) / s + 1;
        long M = (long)N * OH * OW;
        int KTOT = KH * KW * Cin;
        int nChunks = (KTOT + 31) / 32;
        if ((Cin & 31) == 0) {
            long tiles = ((M + 63) / 64) * (Cout / 64);   // 64x64 tile per wave
            int nblk = (int)((tiles + 7) / 8);
            conv_fast_kernel<<<nblk, 256, 0, stream>>>(
                in, (const uint4*)wp, out, N, H, W, Cin, OH, OW, Cout,
                KH, KW, s, pad, nChunks);
        } else {
            long tiles = ((M + 15) / 16) * (Cout / 16);   // 16x16 tile per wave
            int nblk = (int)((tiles + 7) / 8);
            conv1_kernel<<<nblk, 256, 0, stream>>>(
                in, (const uint4*)wp, out, N, H, W, OH, OW, Cout);
        }
    };
    auto bn = [&](const float* src, long M, int C, const float* g, const float* b,
                  const unsigned short* res, int relu, unsigned short* out) {
        bn_stats_kernel<<<C, 256, 0, stream>>>(src, M, C, d_mean, d_var);
        long total = M * C;
        bn_apply_kernel<<<(int)((total + 255) / 256), 256, 0, stream>>>(
            src, d_mean, d_var, g, b, res, relu, out, total, C);
    };
    auto aggregate = [&](unsigned short* act, int H, int W, int C) {
        long HWC = (long)H * W * C;
        hipMemsetAsync(d_csum, 0, (size_t)NC * HWC * sizeof(float), stream);
        agg_mean_kernel<<<(int)((HWC + 255) / 256), 256, 0, stream>>>(act, d_mbuf, HWC, E);
        long ts = 2L * E * HWC;
        agg_scatter_kernel<<<(int)((ts + 255) / 256), 256, 0, stream>>>(act, pairs, d_csum, HWC, 2 * E);
        long ta = (long)E * HWC;
        agg_apply_kernel<<<(int)((ta + 255) / 256), 256, 0, stream>>>(act, d_csum, d_cnt, d_mbuf, ec, HWC, E);
    };

    // ---- forward pass -------------------------------------------------------
    agg_count_kernel<<<1, 64, 0, stream>>>(pairs, 2 * E, d_cnt, NC);

    { // input NCHW f32 -> NHWC bf16
        long total = (long)E * 5 * 256 * 256;
        nchw2nhwc_kernel<<<(int)((total + 255) / 256), 256, 0, stream>>>(
            x_nchw, d_xin, E, 5, 256, 256);
    }

    int OH, OW;
    conv(d_xin, W_conv1, d_cf32, E, 256, 256, 5, 64, 7, 7, 2, 3, OH, OW);   // 128x128
    bn(d_cf32, (long)E * OH * OW, 64, P_g1, P_b1, nullptr, 1, d_actA);
    { // maxpool 3x3 s2 p1 -> 64x64
        long total = (long)E * 64 * 64 * 64;
        maxpool_kernel<<<(int)((total + 255) / 256), 256, 0, stream>>>(
            d_actA, d_actB, E, 128, 128, 64, 64, 64);
    }

    unsigned short* bufs[3] = { d_actB, d_actC, d_actA };
    int curi = 0;
    int curH = 64, curW = 64, curC = 64;

    int bidx = 0;
    for (int L = 0; L < 4; ++L) {
        int nblocks = SPECS[L][1];
        for (int bi = 0; bi < nblocks; ++bi) {
            BP& bp = blocks[bidx++];
            unsigned short* cur  = bufs[curi];
            unsigned short* tmp  = bufs[(curi + 1) % 3];
            unsigned short* outb = bufs[(curi + 2) % 3];
            int oh, ow;
            conv(cur, bp.w1, d_cf32, E, curH, curW, curC, bp.planes,
                 3, 3, bp.stride, 1, oh, ow);
            bn(d_cf32, (long)E * oh * ow, bp.planes, bp.g1, bp.b1, nullptr, 1, tmp);

            const unsigned short* res = cur;
            if (bp.down) {
                int dh, dw;
                conv(cur, bp.dw, d_cf32, E, curH, curW, curC, bp.planes,
                     1, 1, bp.stride, 0, dh, dw);
                bn(d_cf32, (long)E * dh * dw, bp.planes, bp.dg, bp.db, nullptr, 0, d_actR);
                res = d_actR;
            }

            int oh2, ow2;
            conv(tmp, bp.w2, d_cf32, E, oh, ow, bp.planes, bp.planes,
                 3, 3, 1, 1, oh2, ow2);
            bn(d_cf32, (long)E * oh2 * ow2, bp.planes, bp.g2, bp.b2, res, 1, outb);

            curi = (curi + 2) % 3;
            curH = oh2; curW = ow2; curC = bp.planes;
        }
        aggregate(bufs[curi], curH, curW, curC);
    }

    // ---- head ---------------------------------------------------------------
    float* out_pred = (float*)d_out;          // [64]
    float* out_y    = (float*)d_out + E;      // [64,128]
    avgpool_kernel<<<(E * 512 + 255) / 256, 256, 0, stream>>>(
        bufs[curi], d_pool, E, curH * curW, 512);
    fc1_kernel<<<(E * 128 + 255) / 256, 256, 0, stream>>>(
        d_pool, P_fc1w, P_fc1b, out_y, E, 512, 128);
    fc2_kernel<<<1, 64, 0, stream>>>(out_y, P_fc2w, P_fc2b, out_pred, E, 128);

    (void)in_sizes; (void)n_in; (void)out_size; (void)ws_size;
}
